// ToyGeneralizedPotential_90048284328112
// MI455X (gfx1250) — compile-verified, matrix operand order checked by reading, not run on hardware
//
#include <hip/hip_runtime.h>
#include <math.h>

// ---------------------------------------------------------------------------
// ToyGeneralizedPotential on MI455X (gfx1250, wave32)
//
// Pair channel (dominant: ~27 GFLOP) runs on v_wmma_f32_16x16x32_f16.
// Each wave owns 16 pairs (M dim), stages the padded 64-wide feature rows in a
// private LDS buffer, and runs both 64-wide hidden layers as 4 N-tiles x
// 2 K-steps of WMMA with f32 accumulation. Layer-0 bias is folded into the
// GEMM via a constant-1.0 feature at K=37 whose weight row is pb0. SiLU uses
// gfx1250's v_tanh_f32 (1 trans op) via silu(x) = y + y*tanh(y), y = x/2,
// with layer-1 bias folded into the y FMA. Wave-local LDS ordering uses
// inline `s_wait_dscnt 0` so waves never need a block barrier.
// ---------------------------------------------------------------------------

typedef __attribute__((ext_vector_type(16))) _Float16 v16h;
typedef __attribute__((ext_vector_type(8)))  _Float16 h8;
typedef __attribute__((ext_vector_type(8)))  float    v8f;

#define BB 4
#define NN 1024
#define EE 16
#define HH 64
#define PAIRS ((NN * (NN - 1)) / 2)      // 523776, divisible by 16
#define TILES_PER_BATCH (PAIRS / 16)     // 32736
#define PAIR_BLOCKS 256                  // x 8 waves = 2048 waves
#define WAVES_PER_BATCH ((PAIR_BLOCKS * 8) / BB)  // 512

// SiLU given y = x/2 (lets callers fold bias into the halving FMA):
//   silu(x) = x * sigmoid(x) = y * (1 + tanh(y)) = y + y*tanh(y)
// gfx1250 path: 1x v_tanh_f32 + 1x v_fma. Fallback: exp+rcp formulation.
__device__ __forceinline__ float silu_prehalf(float y) {
#if __has_builtin(__builtin_amdgcn_tanhf)
  const float t = __builtin_amdgcn_tanhf(y);
  return fmaf(y, t, y);
#else
  const float x = y + y;
  return x * __builtin_amdgcn_rcpf(1.0f + __expf(-x));
#endif
}

__device__ __forceinline__ float silu_f(float x) {
  return silu_prehalf(0.5f * x);
}

// ---------------------------------------------------------------------------
// Prep: center positions per batch, compute |E| per batch.
// ---------------------------------------------------------------------------
__global__ void prep_kernel(const float* __restrict__ pos,
                            const float* __restrict__ ef,
                            float* __restrict__ cp,
                            float* __restrict__ enorm) {
  const int b = blockIdx.x;
  const int t = threadIdx.x;
  __shared__ float sx[256], sy[256];
  float ax = 0.0f, ay = 0.0f;
  for (int n = t; n < NN; n += 256) {
    ax += pos[(b * NN + n) * 2 + 0];
    ay += pos[(b * NN + n) * 2 + 1];
  }
  sx[t] = ax; sy[t] = ay;
  __syncthreads();
  for (int s = 128; s > 0; s >>= 1) {
    if (t < s) { sx[t] += sx[t + s]; sy[t] += sy[t + s]; }
    __syncthreads();
  }
  const float mx = sx[0] * (1.0f / NN);
  const float my = sy[0] * (1.0f / NN);
  for (int n = t; n < NN; n += 256) {
    cp[(b * NN + n) * 2 + 0] = pos[(b * NN + n) * 2 + 0] - mx;
    cp[(b * NN + n) * 2 + 1] = pos[(b * NN + n) * 2 + 1] - my;
  }
  if (t == 0) {
    const float ex = ef[b * 2 + 0], ey = ef[b * 2 + 1];
    enorm[b] = sqrtf(ex * ex + ey * ey);
  }
}

// ---------------------------------------------------------------------------
// Atom channel: 4096 atoms, 21->64->64->1 MLP. Negligible FLOPs -> plain VALU.
// ---------------------------------------------------------------------------
__global__ void atom_kernel(const float* __restrict__ cp,
                            const int* __restrict__ species,
                            const float* __restrict__ ef,
                            const float* __restrict__ enorm,
                            const float* __restrict__ embed,
                            const float* __restrict__ aW0, const float* __restrict__ ab0,
                            const float* __restrict__ aW1, const float* __restrict__ ab1,
                            const float* __restrict__ aW2, const float* __restrict__ ab2,
                            float* __restrict__ out) {
  const int gid = blockIdx.x * blockDim.x + threadIdx.x;
  if (gid >= BB * NN) return;
  const int b = gid / NN;

  float f[21];
  f[0] = cp[gid * 2 + 0];
  f[1] = cp[gid * 2 + 1];
  f[2] = ef[b * 2 + 0];
  f[3] = ef[b * 2 + 1];
  f[4] = enorm[b];
  const int sp = species[gid];
#pragma unroll
  for (int q = 0; q < EE; ++q) f[5 + q] = embed[sp * EE + q];

  float h0[HH];
#pragma unroll
  for (int o = 0; o < HH; ++o) {
    float t = ab0[o];
#pragma unroll
    for (int k = 0; k < 21; ++k) t += f[k] * aW0[k * HH + o];
    h0[o] = silu_f(t);
  }
  float e = ab2[0];
  for (int o = 0; o < HH; ++o) {
    float t = ab1[o];
#pragma unroll
    for (int k = 0; k < HH; ++k) t += h0[k] * aW1[k * HH + o];
    e += silu_f(t) * aW2[o];
  }
  atomicAdd(&out[b], e);
}

// ---------------------------------------------------------------------------
// WMMA fragment helpers (CDNA5 ISA §7.12.2, 16-bit A 16x32 layout):
//   lane l: row/col = l&15, half = l>>4
//   element e in v16h: K = (e>>3)*16 + half*8 + (e&7)   (+ kblk*32)
// B fragments use the same per-lane K striping with lane -> output column
// (weights loaded "transposed": one column of W per lane).
// ---------------------------------------------------------------------------
__device__ __forceinline__ v16h load_a_frag(const _Float16* row, int half, int kblk) {
  v16h a;
  const _Float16* p = row + kblk * 32 + half * 8;
  ((h8*)&a)[0] = *(const h8*)(p);        // K = kblk*32 + half*8 + 0..7
  ((h8*)&a)[1] = *(const h8*)(p + 16);   // K = kblk*32 + 16 + half*8 + 0..7
  return a;
}

// Layer-0 B-fragment: rows 0..36 from pW0, row 37 = pb0 (bias folded into the
// GEMM against the constant-1.0 feature), rows 38..63 zero.
__device__ __forceinline__ v16h load_b_frag_l0(const float* __restrict__ W,
                                               const float* __restrict__ bias,
                                               int ncol, int kblk, int half) {
  v16h v;
#pragma unroll
  for (int e = 0; e < 16; ++e) {
    const int K = kblk * 32 + (e >> 3) * 16 + half * 8 + (e & 7);
    float x = 0.0f;
    if (K < 37)       x = W[K * HH + ncol];
    else if (K == 37) x = bias[ncol];
    v[e] = (_Float16)x;
  }
  return v;
}

__device__ __forceinline__ v16h load_b_frag(const float* __restrict__ W,
                                            int ncol, int kblk, int half) {
  v16h v;
#pragma unroll
  for (int e = 0; e < 16; ++e) {
    const int K = kblk * 32 + (e >> 3) * 16 + half * 8 + (e & 7);
    v[e] = (_Float16)W[K * HH + ncol];
  }
  return v;
}

// ---------------------------------------------------------------------------
// Pair channel: 37->64->64->1 MLP over 523776 pairs/batch via f16 WMMA.
// ---------------------------------------------------------------------------
__global__ void __launch_bounds__(256, 1) pair_kernel(
    const float* __restrict__ cp, const int* __restrict__ species,
    const float* __restrict__ ef, const float* __restrict__ enorm,
    const float* __restrict__ embed,
    const float* __restrict__ pW0, const float* __restrict__ pb0,
    const float* __restrict__ pW1, const float* __restrict__ pb1,
    const float* __restrict__ pW2, const float* __restrict__ pb2,
    float* __restrict__ out) {
  // per-wave staging: 16 rows x 64 features (stride 72 halves; rows 16B-aligned)
  __shared__ __align__(16) _Float16 smem[8][16][72];

  const int lane = threadIdx.x & 31;
  const int wv = threadIdx.x >> 5;
  const int m = lane & 15;        // A-matrix row / B-matrix column sub-index
  const int half = lane >> 4;     // K-half selector
  const int waveGlobal = blockIdx.x * 8 + wv;
  const int b = waveGlobal & (BB - 1);         // wave pinned to one batch
  const int waveInBatch = waveGlobal >> 2;

  // Preload weight fragments (f32 -> f16 once per wave, live in VGPRs).
  v16h w0f[4][2], w1f[4][2];
  float hb1l[4], w2l[4];
#pragma unroll
  for (int t = 0; t < 4; ++t) {
    const int ncol = t * 16 + m;
#pragma unroll
    for (int kb = 0; kb < 2; ++kb) {
      w0f[t][kb] = load_b_frag_l0(pW0, pb0, ncol, kb, half);
      w1f[t][kb] = load_b_frag(pW1, ncol, kb, half);
    }
    hb1l[t] = 0.5f * pb1[ncol];   // pre-halved layer-1 bias for the SiLU FMA
    w2l[t] = pW2[ncol];
  }
  const float ex = ef[b * 2 + 0], ey = ef[b * 2 + 1];
  const float en = enorm[b];
  const float* cpb = cp + b * NN * 2;
  const int* spb = species + b * NN;

  float laneAcc = 0.0f;
  int ntiles = 0;

  for (int tile = waveInBatch; tile < TILES_PER_BATCH; tile += WAVES_PER_BATCH) {
    ++ntiles;
    const int p = tile * 16 + m;

    // invert flattened triu index -> (i, j), i < j
    const float bq = (float)(2 * NN - 1);
    int i = (int)((bq - sqrtf(bq * bq - 8.0f * (float)p)) * 0.5f);
    if (i < 0) i = 0;
    if (i > NN - 2) i = NN - 2;
    while (i > 0 && (i * NN - (i * (i + 1)) / 2) > p) --i;
    while (((i + 1) * NN - ((i + 1) * (i + 2)) / 2) <= p) ++i;
    const int j = p - (i * NN - (i * (i + 1)) / 2) + i + 1;

    const float xi = cpb[i * 2 + 0], yi = cpb[i * 2 + 1];
    const float xj = cpb[j * 2 + 0], yj = cpb[j * 2 + 1];
    const float rx = xi - xj, ry = yi - yj;
    const float d = sqrtf(rx * rx + ry * ry);
    const float inv = 1.0f / fmaxf(d, 1e-12f);
    const float edotu = ex * (rx * inv) + ey * (ry * inv);
    const int si = spb[i], sj = spb[j];

    float f[37];
    f[0] = rx; f[1] = ry; f[2] = d; f[3] = edotu; f[4] = en;
#pragma unroll
    for (int q = 0; q < EE; ++q) {
      f[5 + q] = embed[si * EE + q];
      f[21 + q] = embed[sj * EE + q];
    }

    // stage this lane's half of row m; K=37 is the constant-1.0 bias feature
    _Float16* row = &smem[wv][m][0];
    if (half == 0) {
#pragma unroll
      for (int c = 0; c < 32; ++c) row[c] = (_Float16)f[c];
    } else {
#pragma unroll
      for (int c = 32; c < 37; ++c) row[c] = (_Float16)f[c];
      row[37] = (_Float16)1.0f;
#pragma unroll
      for (int c = 38; c < 64; ++c) row[c] = (_Float16)0.0f;
    }
    asm volatile("s_wait_dscnt 0" ::: "memory");  // wave-local LDS visibility

    v16h a0 = load_a_frag(row, half, 0);
    v16h a1 = load_a_frag(row, half, 1);

    // ---- layer 0 (bias inside GEMM): SiLU, restage as f16 (transpose via LDS)
#pragma unroll
    for (int t = 0; t < 4; ++t) {
      v8f acc = {};
      acc = __builtin_amdgcn_wmma_f32_16x16x32_f16(false, a0, false, w0f[t][0],
                                                   (short)0, acc, false, false);
      acc = __builtin_amdgcn_wmma_f32_16x16x32_f16(false, a1, false, w0f[t][1],
                                                   (short)0, acc, false, false);
      // C layout: lanes 0-15 -> M=r, lanes 16-31 -> M=r+8; N = t*16 + (lane&15)
#pragma unroll
      for (int r = 0; r < 8; ++r) {
        smem[wv][r + 8 * half][t * 16 + m] = (_Float16)silu_prehalf(0.5f * acc[r]);
      }
    }
    asm volatile("s_wait_dscnt 0" ::: "memory");

    a0 = load_a_frag(row, half, 0);
    a1 = load_a_frag(row, half, 1);

    // ---- layer 1 + fused final layer: sum over all C elements of silu(.)*w2[N]
#pragma unroll
    for (int t = 0; t < 4; ++t) {
      v8f acc = {};
      acc = __builtin_amdgcn_wmma_f32_16x16x32_f16(false, a0, false, w1f[t][0],
                                                   (short)0, acc, false, false);
      acc = __builtin_amdgcn_wmma_f32_16x16x32_f16(false, a1, false, w1f[t][1],
                                                   (short)0, acc, false, false);
#pragma unroll
      for (int r = 0; r < 8; ++r) {
        // y = (acc + b1)/2 via one FMA, then silu = y + y*tanh(y)
        laneAcc += silu_prehalf(fmaf(0.5f, acc[r], hb1l[t])) * w2l[t];
      }
    }
  }

  // per-pair output bias (16 pairs per tile handled by this wave)
  if (lane == 0) laneAcc += (float)ntiles * 16.0f * pb2[0];

  // wave32 tree reduction, one atomic per wave
#pragma unroll
  for (int o = 16; o > 0; o >>= 1) laneAcc += __shfl_xor(laneAcc, o, 32);
  if (lane == 0) atomicAdd(&out[b], laneAcc);
}

// ---------------------------------------------------------------------------
extern "C" void kernel_launch(void* const* d_in, const int* in_sizes, int n_in,
                              void* d_out, int out_size, void* d_ws, size_t ws_size,
                              hipStream_t stream) {
  const float* positions = (const float*)d_in[0];
  const int*   species   = (const int*)d_in[1];
  const float* efield    = (const float*)d_in[2];
  const float* embed     = (const float*)d_in[3];
  const float* aW0 = (const float*)d_in[4];
  const float* ab0 = (const float*)d_in[5];
  const float* aW1 = (const float*)d_in[6];
  const float* ab1 = (const float*)d_in[7];
  const float* aW2 = (const float*)d_in[8];
  const float* ab2 = (const float*)d_in[9];
  const float* pW0 = (const float*)d_in[10];
  const float* pb0 = (const float*)d_in[11];
  const float* pW1 = (const float*)d_in[12];
  const float* pb1 = (const float*)d_in[13];
  const float* pW2 = (const float*)d_in[14];
  const float* pb2 = (const float*)d_in[15];

  float* out = (float*)d_out;
  float* cp = (float*)d_ws;             // [B, N, 2] centered positions
  float* enorm = cp + BB * NN * 2;      // [B]

  hipMemsetAsync(d_out, 0, (size_t)out_size * sizeof(float), stream);
  prep_kernel<<<BB, 256, 0, stream>>>(positions, efield, cp, enorm);
  atom_kernel<<<(BB * NN) / 256, 256, 0, stream>>>(cp, species, efield, enorm, embed,
                                                   aW0, ab0, aW1, ab1, aW2, ab2, out);
  pair_kernel<<<PAIR_BLOCKS, 256, 0, stream>>>(cp, species, efield, enorm, embed,
                                               pW0, pb0, pW1, pb1, pW2, pb2, out);
}
